// Bert_44452911514066
// MI455X (gfx1250) — matrix operand order repaired and checked by reading, not compiled, not run on hardware
//
#include <hip/hip_runtime.h>
#include <hip/hip_bf16.h>

typedef __bf16 bf16;
typedef __attribute__((ext_vector_type(16))) __bf16 v16bf;
typedef __attribute__((ext_vector_type(8)))  __bf16 v8bf;
typedef __attribute__((ext_vector_type(8)))  float  v8f;
typedef __attribute__((ext_vector_type(4)))  int    v4i;

#define H_DIM   768
#define NH      12
#define HD      64
#define SEQ     512
#define BATCH   16
#define SB      (SEQ*BATCH)     /* 8192 */
#define NPOS    63
#define NPOSP   64
#define LN_EPS  1e-7f
#define QSCALE  0.07216878364870322f   /* 1/sqrt(3*64) */
#define KSTEPS  (H_DIM/32)             /* 24 */
#define BT_LD   40                     /* padded LDS row: 80B, conflict-free */

#if __has_builtin(__builtin_amdgcn_global_load_async_to_lds_b128)
#define ASYNC_LDS 1
#endif

// ---------------------------------------------------------------------------
// WMMA fragment helpers (CDNA5 wave32, 16x16x32 bf16, f32 accum)
// A (MxK = 16x32): lane={half,m}; elems K = half*16+[0..15].
// B (KxN = 32x16): lane={half,n}; elems K = half*16+[0..15] of column n
//                  (B stored transposed: [N][K], contiguous along K).
// C/D (16x16 f32): elem v -> M = v + 8*half, N = lane&15.
// ---------------------------------------------------------------------------
__device__ __forceinline__ v16bf loadA16(const bf16* base, int ld, int k0) {
    int lane = threadIdx.x & 31;
    int half = lane >> 4, m = lane & 15;
    const bf16* p = base + m * ld + k0 + half * 16;
    v8bf lo = *(const v8bf*)(p);
    v8bf hi = *(const v8bf*)(p + 8);
    v16bf r;
#pragma unroll
    for (int i = 0; i < 8; ++i) { r[i] = lo[i]; r[i + 8] = hi[i]; }
    return r;
}

__device__ __forceinline__ v16bf loadB16(const bf16* baseT, int ld, int k0) {
    int lane = threadIdx.x & 31;
    int half = lane >> 4, n = lane & 15;
    return *(const v16bf*)(baseT + n * ld + k0 + half * 16);
}

// 2x16B variant (16B alignment only; used for padded LDS tiles)
__device__ __forceinline__ v16bf loadB16_2x8(const bf16* baseT, int ld, int k0) {
    int lane = threadIdx.x & 31;
    int half = lane >> 4, n = lane & 15;
    const bf16* p = baseT + n * ld + k0 + half * 16;
    v8bf lo = *(const v8bf*)(p);
    v8bf hi = *(const v8bf*)(p + 8);
    v16bf r;
#pragma unroll
    for (int i = 0; i < 8; ++i) { r[i] = lo[i]; r[i + 8] = hi[i]; }
    return r;
}

__device__ __forceinline__ v8f wmma_bf16(v16bf a, v16bf b, v8f c) {
    return __builtin_amdgcn_wmma_f32_16x16x32_bf16(
        false, a, false, b, (short)0, c, false, false);
}

// ---------------------------------------------------------------------------
// Async global->LDS staging (CDNA5 ASYNCcnt path), with portable fallback.
// Builtin signature (from clang diagnostic): param0 = v4i in AS(1) (non-const),
// param1 = LDS v4i in AS(3), then imm offset, imm cpol.
// Generic LDS pointer low 32 bits == LDS byte offset (ISA 10.2 flat->LDS).
// ---------------------------------------------------------------------------
__device__ __forceinline__ void async_copy_b128(const bf16* src, bf16* dst) {
#ifdef ASYNC_LDS
    typedef __attribute__((address_space(1))) v4i gv4i;
    typedef __attribute__((address_space(3))) v4i lv4i;
    __builtin_amdgcn_global_load_async_to_lds_b128(
        (gv4i*)(unsigned long long)src,
        (lv4i*)(unsigned int)(unsigned long long)dst,
        0, 0);
#else
    *(v8bf*)dst = *(const v8bf*)src;
#endif
}

__device__ __forceinline__ void async_wait0() {
#ifdef ASYNC_LDS
#if __has_builtin(__builtin_amdgcn_s_wait_asynccnt)
    __builtin_amdgcn_s_wait_asynccnt(0);
#else
    asm volatile("s_wait_asynccnt 0x0" ::: "memory");
#endif
#endif
}

// ---------------------------------------------------------------------------
__global__ __launch_bounds__(256)
void cvt_bf16(const float* __restrict__ src, bf16* __restrict__ dst, int n) {
    int i = blockIdx.x * 256 + threadIdx.x;
    if (i < n) dst[i] = (bf16)src[i];
}

// ---------------------------------------------------------------------------
// Row LayerNorm over H=768, optional gate (x*g), bf16 output.
// ---------------------------------------------------------------------------
__global__ __launch_bounds__(256)
void ln_rows(const float* __restrict__ x, const float* __restrict__ g,
             int use_gate, bf16* __restrict__ out) {
    int row = blockIdx.x;
    const float* xr = x + (size_t)row * H_DIM;
    float loc[3];
    float s = 0.f, s2 = 0.f;
#pragma unroll
    for (int i = 0; i < 3; ++i) {
        int c = threadIdx.x + i * 256;
        float v = xr[c];
        if (use_gate) v *= g[(size_t)row * H_DIM + c];
        loc[i] = v; s += v; s2 += v * v;
    }
    __shared__ float rs[256], rq[256];
    rs[threadIdx.x] = s; rq[threadIdx.x] = s2;
    __syncthreads();
    for (int off = 128; off > 0; off >>= 1) {
        if (threadIdx.x < off) {
            rs[threadIdx.x] += rs[threadIdx.x + off];
            rq[threadIdx.x] += rq[threadIdx.x + off];
        }
        __syncthreads();
    }
    float mean = rs[0] * (1.0f / H_DIM);
    float var  = fmaxf(rq[0] * (1.0f / H_DIM) - mean * mean, 0.f);
    float inv  = rsqrtf(var + LN_EPS);
#pragma unroll
    for (int i = 0; i < 3; ++i) {
        int c = threadIdx.x + i * 256;
        out[(size_t)row * H_DIM + c] = (bf16)((loc[i] - mean) * inv);
    }
}

// ---------------------------------------------------------------------------
// Relative-position projection: pos = rel[63,768] @ Wqk^T + b.
// q_pos gets QSCALE folded in; outputs [NH][64(pad)][HD] bf16.
// ---------------------------------------------------------------------------
__global__ __launch_bounds__(256)
void pos_proj(const float* __restrict__ rel, const float* __restrict__ wqk,
              const float* __restrict__ bqk,
              bf16* __restrict__ qposT, bf16* __restrict__ kposT) {
    int row = blockIdx.x;                 // 0..62
    __shared__ float rsh[H_DIM];
    for (int i = threadIdx.x; i < H_DIM; i += 256) rsh[i] = rel[(size_t)row * H_DIM + i];
    __syncthreads();
    for (int n = threadIdx.x; n < 2 * H_DIM; n += 256) {
        const float* wr = wqk + (size_t)n * H_DIM;
        float acc = bqk[n];
        for (int k = 0; k < H_DIM; ++k) acc += rsh[k] * wr[k];
        if (n < H_DIM) {
            int hd = n >> 6, d = n & 63;
            qposT[((size_t)hd * NPOSP + row) * HD + d] = (bf16)(acc * QSCALE);
        } else {
            int n2 = n - H_DIM; int hd = n2 >> 6, d = n2 & 63;
            kposT[((size_t)hd * NPOSP + row) * HD + d] = (bf16)acc;
        }
    }
}

// ---------------------------------------------------------------------------
// WMMA GEMM Y[M,N]=A[M,768]@BT[N,768]^T + bias, mode-based scatter.
// Block = 8 waves x (16x64) = 128x64 tile. The 64x32 B tile is staged in LDS
// once per block via double-buffered async global->LDS copies; A fragments are
// register double-buffered so loads overlap WMMAs.
// mode 0: q (pre-scaled) / k -> [B,NH,S,HD] bf16
// mode 1: v -> [B,NH,HD,S] bf16 (transposed), g -> exact-GELU f32 [S,B,H]
// mode 2: f32 row-major [M,768] (== [S,B,H])
// ---------------------------------------------------------------------------
__global__ __launch_bounds__(256)
void gemm_scatter(const bf16* __restrict__ A, const bf16* __restrict__ BT,
                  const float* __restrict__ bias, int mode,
                  bf16* __restrict__ o_q, bf16* __restrict__ o_k,
                  bf16* __restrict__ o_vt, float* __restrict__ o_g,
                  float* __restrict__ o_y) {
    __shared__ bf16 Bt[2][64][BT_LD];   // 2 x 64 rows x 32(+8 pad) bf16

    int wave = threadIdx.x >> 5;
    int lane = threadIdx.x & 31;
    int half = lane >> 4, ln = lane & 15;
    int row0 = blockIdx.x * 128 + wave * 16;
    int col0 = blockIdx.y * 64;
    const bf16* Ab = A + (size_t)row0 * H_DIM;

    // staging map: thread -> (row n of tile, 8-element chunk)
    int sn = threadIdx.x >> 2;
    int sc8 = (threadIdx.x & 3) * 8;
    const bf16* srcB = BT + (size_t)(col0 + sn) * H_DIM + sc8;

    async_copy_b128(srcB, &Bt[0][sn][sc8]);          // stage k-step 0
    v16bf a_cur = loadA16(Ab, H_DIM, 0);

    v8f acc[4] = {};
    for (int ks = 0; ks < KSTEPS; ++ks) {
        int buf = ks & 1;
        async_wait0();
        __syncthreads();                              // staged tile visible
        if (ks + 1 < KSTEPS) {
            async_copy_b128(srcB + (ks + 1) * 32, &Bt[buf ^ 1][sn][sc8]);
#ifndef ASYNC_LDS
            __builtin_prefetch(srcB + (ks + 2) * 32, 0, 1);
#endif
        }
        v16bf a_nxt = a_cur;
        if (ks + 1 < KSTEPS) a_nxt = loadA16(Ab, H_DIM, (ks + 1) * 32);

        v16bf bfr[4];
#pragma unroll
        for (int t = 0; t < 4; ++t)
            bfr[t] = loadB16_2x8(&Bt[buf][t * 16][0], BT_LD, 0);
#pragma unroll
        for (int t = 0; t < 4; ++t)
            acc[t] = wmma_bf16(a_cur, bfr[t], acc[t]);

        a_cur = a_nxt;
        __syncthreads();                              // all reads of buf done
    }

#pragma unroll
    for (int t = 0; t < 4; ++t) {
        int n = col0 + t * 16 + ln;
        float bv = bias[n];
#pragma unroll
        for (int v = 0; v < 8; ++v) {
            int m = row0 + v + 8 * half;
            float val = acc[t][v] + bv;
            int s = m >> 4, b = m & 15;          // m = s*BATCH + b
            if (mode == 0) {
                if (n < H_DIM) {
                    int hd = n >> 6, d = n & 63;
                    o_q[(((size_t)b * NH + hd) * SEQ + s) * HD + d] = (bf16)(val * QSCALE);
                } else {
                    int n2 = n - H_DIM; int hd = n2 >> 6, d = n2 & 63;
                    o_k[(((size_t)b * NH + hd) * SEQ + s) * HD + d] = (bf16)val;
                }
            } else if (mode == 1) {
                if (n < H_DIM) {
                    int hd = n >> 6, d = n & 63;
                    o_vt[(((size_t)b * NH + hd) * HD + d) * SEQ + s] = (bf16)val;
                } else {
                    int n2 = n - H_DIM;
                    float gl = 0.5f * val * (1.0f + erff(val * 0.7071067811865475f));
                    o_g[((size_t)s * BATCH + b) * H_DIM + n2] = gl;
                }
            } else {
                o_y[(size_t)m * H_DIM + n] = val;
            }
        }
    }
}

// ---------------------------------------------------------------------------
// Bucket score tables: cq = (q*SCALE).k_pos, ck = k.(q_pos*SCALE), f32 [bh,S,64].
// ---------------------------------------------------------------------------
__global__ __launch_bounds__(256)
void cqck(const bf16* __restrict__ qb, const bf16* __restrict__ kb,
          const bf16* __restrict__ qposT, const bf16* __restrict__ kposT,
          float* __restrict__ cq, float* __restrict__ ck) {
    int wave = threadIdx.x >> 5, lane = threadIdx.x & 31;
    int half = lane >> 4, ln = lane & 15;
    int bh = blockIdx.x >> 2, part = blockIdx.x & 3;
    int head = bh % NH;
    int row0 = part * 128 + wave * 16;

#pragma unroll
    for (int which = 0; which < 2; ++which) {
        const bf16* Ab = (which == 0 ? qb : kb) + ((size_t)bh * SEQ + row0) * HD;
        const bf16* Bb = (which == 0 ? kposT : qposT) + (size_t)head * NPOSP * HD;
        float* outp = (which == 0 ? cq : ck);
        v16bf a0 = loadA16(Ab, HD, 0);
        v16bf a1 = loadA16(Ab, HD, 32);
        v16bf bfr[8];
#pragma unroll
        for (int t = 0; t < 4; ++t) {
            bfr[2 * t]     = loadB16(Bb + (size_t)t * 16 * HD, HD, 0);
            bfr[2 * t + 1] = loadB16(Bb + (size_t)t * 16 * HD, HD, 32);
        }
#pragma unroll
        for (int t = 0; t < 4; ++t) {
            v8f c = {};
            c = wmma_bf16(a0, bfr[2 * t], c);
            c = wmma_bf16(a1, bfr[2 * t + 1], c);
#pragma unroll
            for (int v = 0; v < 8; ++v)
                outp[((size_t)bh * SEQ + row0 + v + 8 * half) * NPOSP + t * 16 + ln] = c[v];
        }
    }
}

// ---------------------------------------------------------------------------
// Attention: block per (b, head, 16-query tile).
// A: WMMA Q.K^T -> LDS scores (B fragments preloaded to overlap loads/WMMA).
// B: gather cq/ck via position_indices + key mask. C: masked softmax.
// D: WMMA P.V, per-wave 64-key split, ds_add_f32 reduction, write [S,B,H].
// ---------------------------------------------------------------------------
__global__ __launch_bounds__(256)
void attention(const bf16* __restrict__ qb, const bf16* __restrict__ kb,
               const bf16* __restrict__ vtb,
               const float* __restrict__ cq, const float* __restrict__ ck,
               const int* __restrict__ pidx, const unsigned char* __restrict__ amask,
               float* __restrict__ ctxg) {
    __shared__ float sc[16][520];
    __shared__ bf16  pr[16][520];
    __shared__ float ctxt[16][64];
    __shared__ float red[16][17];

    int wave = threadIdx.x >> 5, lane = threadIdx.x & 31;
    int half = lane >> 4, ln = lane & 15;
    int bh = blockIdx.x >> 5;
    int q0 = (blockIdx.x & 31) * 16;
    int b = bh / NH, head = bh % NH;

    const bf16* vrow = vtb + (size_t)bh * HD * SEQ;   // [HD][SEQ]
    // pull the 64KB V tile toward the caches while we do scores/softmax
    __builtin_prefetch(vrow + (size_t)threadIdx.x * 128, 0, 1);

    for (int i = threadIdx.x; i < 16 * 64; i += 256) ctxt[i >> 6][i & 63] = 0.f;

    // ---- Phase A: content-content scores (q pre-scaled) ----
    const bf16* qrow = qb + ((size_t)bh * SEQ + q0) * HD;
    const bf16* krow = kb + (size_t)bh * SEQ * HD;
    v16bf a0 = loadA16(qrow, HD, 0);
    v16bf a1 = loadA16(qrow, HD, 32);
    {
        v16bf bfr[8];
#pragma unroll
        for (int t = 0; t < 4; ++t) {
            int ncol = wave * 64 + t * 16;
            bfr[2 * t]     = loadB16(krow + (size_t)ncol * HD, HD, 0);
            bfr[2 * t + 1] = loadB16(krow + (size_t)ncol * HD, HD, 32);
        }
#pragma unroll
        for (int t = 0; t < 4; ++t) {
            int ncol = wave * 64 + t * 16;
            v8f c = {};
            c = wmma_bf16(a0, bfr[2 * t], c);
            c = wmma_bf16(a1, bfr[2 * t + 1], c);
#pragma unroll
            for (int v = 0; v < 8; ++v) sc[v + 8 * half][ncol + ln] = c[v];
        }
    }
    __syncthreads();

    // ---- Phase B: relative-position gathers + key mask ----
    const float* cqp = cq + (size_t)bh * SEQ * NPOSP;
    const float* ckp = ck + (size_t)bh * SEQ * NPOSP;
    const unsigned char* am = amask + (size_t)b * SEQ;
    for (int i = threadIdx.x; i < 16 * SEQ; i += 256) {
        int qq = i >> 9, kk = i & 511;
        float sv;
        if (am[kk]) {
            sv = -__builtin_inff();
        } else {
            int bucket = pidx[(size_t)(q0 + qq) * SEQ + kk];
            sv = sc[qq][kk] + cqp[(size_t)(q0 + qq) * NPOSP + bucket]
                            + ckp[(size_t)kk * NPOSP + bucket];
        }
        sc[qq][kk] = sv;
    }
    __syncthreads();

    // ---- Phase C: masked softmax (16 threads per row) ----
    int row = threadIdx.x >> 4, seg = threadIdx.x & 15;
    float mx = -__builtin_inff();
    for (int c0 = 0; c0 < 32; ++c0) mx = fmaxf(mx, sc[row][seg * 32 + c0]);
    red[row][seg] = mx;
    __syncthreads();
    if (seg == 0) {
        float m2 = red[row][0];
        for (int j = 1; j < 16; ++j) m2 = fmaxf(m2, red[row][j]);
        red[row][16] = m2;
    }
    __syncthreads();
    float rowmax = red[row][16];
    float sum = 0.f;
    for (int c0 = 0; c0 < 32; ++c0) {
        float e = __expf(sc[row][seg * 32 + c0] - rowmax);  // exp(-inf)=0 on mask
        sc[row][seg * 32 + c0] = e; sum += e;
    }
    __syncthreads();
    red[row][seg] = sum;
    __syncthreads();
    if (seg == 0) {
        float s2 = 0.f;
        for (int j = 0; j < 16; ++j) s2 += red[row][j];
        red[row][16] = s2;
    }
    __syncthreads();
    float inv = 1.0f / red[row][16];
    for (int c0 = 0; c0 < 32; ++c0)
        pr[row][seg * 32 + c0] = (bf16)(sc[row][seg * 32 + c0] * inv);
    __syncthreads();

    // ---- Phase D: P @ V, per-wave 64-key split, LDS f32 reduction ----
    {
        v16bf apd[2];
        apd[0] = loadA16(&pr[0][0], 520, wave * 64);
        apd[1] = loadA16(&pr[0][0], 520, wave * 64 + 32);
        v16bf bfr[8];
#pragma unroll
        for (int t = 0; t < 4; ++t) {
            bfr[2 * t]     = loadB16(vrow + (size_t)(t * 16) * SEQ, SEQ, wave * 64);
            bfr[2 * t + 1] = loadB16(vrow + (size_t)(t * 16) * SEQ, SEQ, wave * 64 + 32);
        }
#pragma unroll
        for (int t = 0; t < 4; ++t) {
            v8f c = {};
            c = wmma_bf16(apd[0], bfr[2 * t], c);
            c = wmma_bf16(apd[1], bfr[2 * t + 1], c);
#pragma unroll
            for (int v = 0; v < 8; ++v)
                atomicAdd(&ctxt[v + 8 * half][t * 16 + ln], c[v]);
        }
    }
    __syncthreads();
    for (int i = threadIdx.x; i < 16 * 64; i += 256) {
        int qq = i >> 6, d = i & 63;
        ctxg[(size_t)(q0 + qq) * (BATCH * H_DIM) + (size_t)b * H_DIM + head * HD + d]
            = ctxt[qq][d];
    }
}

// ---------------------------------------------------------------------------
extern "C" void kernel_launch(void* const* d_in, const int* in_sizes, int n_in,
                              void* d_out, int out_size, void* d_ws, size_t ws_size,
                              hipStream_t stream) {
    const float* hidden = (const float*)d_in[0];   // [S,B,H]
    const float* rel    = (const float*)d_in[1];   // [63,H]
    const float* w_qk   = (const float*)d_in[2];   // [2H,H]
    const float* b_qk   = (const float*)d_in[3];
    const float* w_vg   = (const float*)d_in[4];
    const float* b_vg   = (const float*)d_in[5];
    const float* w_out  = (const float*)d_in[6];   // [H,H]
    const float* b_out  = (const float*)d_in[7];
    const unsigned char* amask = (const unsigned char*)d_in[8]; // bool [B,1,1,S]
    const int*   pidx   = (const int*)d_in[9];     // int32 [S,S]
    float* out = (float*)d_out;

    char* ws = (char*)d_ws;
    size_t off = 0;
    auto alloc = [&](size_t bytes) -> char* {
        char* p = ws + off;
        off = (off + bytes + 255) & ~(size_t)255;
        return p;
    };
    bf16*  hbf    = (bf16*)alloc((size_t)SB * H_DIM * 2);
    bf16*  ctxbf  = (bf16*)alloc((size_t)SB * H_DIM * 2);
    bf16*  wqkbf  = (bf16*)alloc((size_t)2 * H_DIM * H_DIM * 2);
    bf16*  wvgbf  = (bf16*)alloc((size_t)2 * H_DIM * H_DIM * 2);
    bf16*  woutbf = (bf16*)alloc((size_t)H_DIM * H_DIM * 2);
    bf16*  qbfp   = (bf16*)alloc((size_t)BATCH * NH * SEQ * HD * 2);
    bf16*  kbfp   = (bf16*)alloc((size_t)BATCH * NH * SEQ * HD * 2);
    bf16*  vtbfp  = (bf16*)alloc((size_t)BATCH * NH * SEQ * HD * 2);
    float* gf     = (float*)alloc((size_t)SB * H_DIM * 4);
    bf16*  qposT  = (bf16*)alloc((size_t)NH * NPOSP * HD * 2);
    bf16*  kposT  = (bf16*)alloc((size_t)NH * NPOSP * HD * 2);
    float* cqp    = (float*)alloc((size_t)BATCH * NH * SEQ * NPOSP * 4);
    float* ckp    = (float*)alloc((size_t)BATCH * NH * SEQ * NPOSP * 4);
    float* ctxg   = (float*)alloc((size_t)SB * H_DIM * 4);

    // 1) weights -> bf16
    int nw1 = 2 * H_DIM * H_DIM;
    cvt_bf16<<<(nw1 + 255) / 256, 256, 0, stream>>>(w_qk, wqkbf, nw1);
    cvt_bf16<<<(nw1 + 255) / 256, 256, 0, stream>>>(w_vg, wvgbf, nw1);
    int nw2 = H_DIM * H_DIM;
    cvt_bf16<<<(nw2 + 255) / 256, 256, 0, stream>>>(w_out, woutbf, nw2);

    // 2) pre-LN -> bf16 activations
    ln_rows<<<SB, 256, 0, stream>>>(hidden, nullptr, 0, hbf);

    // 3) relative-position projection (pad row zeroed)
    (void)hipMemsetAsync(qposT, 0, (size_t)NH * NPOSP * HD * 2, stream);
    (void)hipMemsetAsync(kposT, 0, (size_t)NH * NPOSP * HD * 2, stream);
    pos_proj<<<NPOS, 256, 0, stream>>>(rel, w_qk, b_qk, qposT, kposT);

    // 4) fused QK / VG projections
    dim3 gqk(SB / 128, (2 * H_DIM) / 64);   // (64, 24)
    gemm_scatter<<<gqk, 256, 0, stream>>>(hbf, wqkbf, b_qk, 0,
                                          qbfp, kbfp, nullptr, nullptr, nullptr);
    gemm_scatter<<<gqk, 256, 0, stream>>>(hbf, wvgbf, b_vg, 1,
                                          nullptr, nullptr, vtbfp, gf, nullptr);

    // 5) bucket score tables
    cqck<<<BATCH * NH * 4, 256, 0, stream>>>(qbfp, kbfp, qposT, kposT, cqp, ckp);

    // 6) attention
    attention<<<BATCH * NH * 32, 256, 0, stream>>>(qbfp, kbfp, vtbfp,
                                                   cqp, ckp, pidx, amask, ctxg);

    // 7) gated post-LN -> bf16
    ln_rows<<<SB, 256, 0, stream>>>(ctxg, gf, 1, ctxbf);

    // 8) output projection -> d_out f32 [S,B,H]
    dim3 gout(SB / 128, H_DIM / 64);        // (64, 12)
    gemm_scatter<<<gout, 256, 0, stream>>>(ctxbf, woutbf, b_out, 2,
                                           nullptr, nullptr, nullptr, nullptr, out);
}
